// TriPruneHGNN_Complete_7352984011021
// MI455X (gfx1250) — 16-bit and 8-bit matrix operand order repaired and checked by reading, compiled, not run on hardware
//
#include <hip/hip_runtime.h>
#include <hip/hip_bf16.h>
#include <math.h>

#define KK 4
#define NN 2048
#define EE 4096
#define DD 128
#define DH 64
#define DX 160
#define BETA_C 0.6f
#define GEPS 0.01f
#define LAM_C 0.05f
#define TC0_C 0.3f
#define TCMAX_C 0.7f
#define MAXSC 500

typedef __attribute__((ext_vector_type(16))) _Float16 v16h;
typedef __attribute__((ext_vector_type(8)))  _Float16 v8h;
typedef __attribute__((ext_vector_type(8)))  float    v8f;
typedef __attribute__((ext_vector_type(8)))  int      v8i;

__device__ __forceinline__ float clamp01(float x) { return fminf(fmaxf(x, 0.f), 1.f); }

__device__ __forceinline__ v16h load_a_f16(const float* __restrict__ row,
                                           int k0, int laneHi)
{
  const float4* p0 = (const float4*)(row + k0 + laneHi * 8);
  const float4* p1 = (const float4*)(row + k0 + 16 + laneHi * 8);
  float4 q0 = p0[0], q1 = p0[1], q2 = p1[0], q3 = p1[1];
  v16h a;
  a[0]=(_Float16)q0.x; a[1]=(_Float16)q0.y; a[2]=(_Float16)q0.z; a[3]=(_Float16)q0.w;
  a[4]=(_Float16)q1.x; a[5]=(_Float16)q1.y; a[6]=(_Float16)q1.z; a[7]=(_Float16)q1.w;
  a[8]=(_Float16)q2.x; a[9]=(_Float16)q2.y; a[10]=(_Float16)q2.z; a[11]=(_Float16)q2.w;
  a[12]=(_Float16)q3.x; a[13]=(_Float16)q3.y; a[14]=(_Float16)q3.z; a[15]=(_Float16)q3.w;
  return a;
}

__device__ __forceinline__ v16h load_b_f16(const _Float16* __restrict__ colrow,
                                           int k0, int laneHi)
{
  const v8h* p = (const v8h*)(colrow + k0 + laneHi * 8);
  union { v16h v; v8h h[2]; } u;
  u.h[0] = p[0]; u.h[1] = p[2];
  return u.v;
}

__device__ __forceinline__ v8i load_adjA(const unsigned char* __restrict__ row,
                                         int laneHi)
{
  int2 q0 = *(const int2*)(row + laneHi * 8);
  int2 q1 = *(const int2*)(row + 16 + laneHi * 8);
  int2 q2 = *(const int2*)(row + 32 + laneHi * 8);
  int2 q3 = *(const int2*)(row + 48 + laneHi * 8);
  v8i a;
  a[0]=q0.x; a[1]=q0.y; a[2]=q1.x; a[3]=q1.y;
  a[4]=q2.x; a[5]=q2.y; a[6]=q3.x; a[7]=q3.y;
  return a;
}

__device__ __forceinline__ v8i load_adjB(const unsigned char* __restrict__ row,
                                         int laneHi)
{
  int2 q0 = *(const int2*)(row + laneHi * 8);
  int2 q1 = *(const int2*)(row + 16 + laneHi * 8);
  int2 q2 = *(const int2*)(row + 32 + laneHi * 8);
  int2 q3 = *(const int2*)(row + 48 + laneHi * 8);
  v8i b;
  b[0]=q0.x; b[1]=q0.y; b[2]=q1.x; b[3]=q1.y;
  b[4]=q2.x; b[5]=q2.y; b[6]=q3.x; b[7]=q3.y;
  return b;
}

__global__ __launch_bounds__(32)
void gemm_f16_wmma(const float* __restrict__ A, int lda,
                   const _Float16* __restrict__ Bt,
                   const float* __restrict__ bias,
                   float* __restrict__ C, int ldc,
                   int Kd, int act)
{
  const int lane = threadIdx.x, laneLo = lane & 15, laneHi = lane >> 4;
  const float* __restrict__ arow0 = A + (size_t)(blockIdx.x * 32 + laneLo) * lda;
  const float* __restrict__ arow1 = arow0 + (size_t)16 * lda;
  const _Float16* __restrict__ bcol0 = Bt + (size_t)(blockIdx.y * 32 + laneLo) * Kd;
  const _Float16* __restrict__ bcol1 = bcol0 + (size_t)16 * Kd;
  v8f acc00 = {}, acc01 = {}, acc10 = {}, acc11 = {};
  for (int k0 = 0; k0 < Kd; k0 += 32) {
    v16h a0 = load_a_f16(arow0, k0, laneHi);
    v16h a1 = load_a_f16(arow1, k0, laneHi);
    v16h b0 = load_b_f16(bcol0, k0, laneHi);
    v16h b1 = load_b_f16(bcol1, k0, laneHi);
    acc00 = __builtin_amdgcn_wmma_f32_16x16x32_f16(false, a0, false, b0, (short)0, acc00, false, false);
    acc01 = __builtin_amdgcn_wmma_f32_16x16x32_f16(false, a0, false, b1, (short)0, acc01, false, false);
    acc10 = __builtin_amdgcn_wmma_f32_16x16x32_f16(false, a1, false, b0, (short)0, acc10, false, false);
    acc11 = __builtin_amdgcn_wmma_f32_16x16x32_f16(false, a1, false, b1, (short)0, acc11, false, false);
  }
  const int col0 = blockIdx.y * 32 + laneLo;
  const int col1 = col0 + 16;
  const float bv0 = bias[col0], bv1 = bias[col1];
#pragma unroll
  for (int r = 0; r < 8; ++r) {
    int row0 = blockIdx.x * 32 + r + 8 * laneHi;
    int row1 = row0 + 16;
    float v00 = acc00[r] + bv0, v01 = acc01[r] + bv1;
    float v10 = acc10[r] + bv0, v11 = acc11[r] + bv1;
    if (act == 1) {
      v00 = fmaxf(v00, 0.f); v01 = fmaxf(v01, 0.f);
      v10 = fmaxf(v10, 0.f); v11 = fmaxf(v11, 0.f);
    }
    C[(size_t)row0 * ldc + col0] = v00;
    C[(size_t)row0 * ldc + col1] = v01;
    C[(size_t)row1 * ldc + col0] = v10;
    C[(size_t)row1 * ldc + col1] = v11;
  }
}

__global__ __launch_bounds__(32)
void adj_wmma_iu8(const unsigned char* __restrict__ Bi8,
                  float* __restrict__ AdjF)
{
  const int lane = threadIdx.x, laneLo = lane & 15, laneHi = lane >> 4;
  const unsigned char* __restrict__ ri0 = Bi8 + (size_t)(blockIdx.x * 32 + laneLo) * EE;
  const unsigned char* __restrict__ ri1 = ri0 + (size_t)16 * EE;
  const unsigned char* __restrict__ rj0 = Bi8 + (size_t)(blockIdx.y * 32 + laneLo) * EE;
  const unsigned char* __restrict__ rj1 = rj0 + (size_t)16 * EE;
  v8i acc00 = {}, acc01 = {}, acc10 = {}, acc11 = {};
  for (int e0 = 0; e0 < EE; e0 += 64) {
    v8i a0 = load_adjA(ri0 + e0, laneHi);
    v8i a1 = load_adjA(ri1 + e0, laneHi);
    v8i b0 = load_adjB(rj0 + e0, laneHi);
    v8i b1 = load_adjB(rj1 + e0, laneHi);
    acc00 = __builtin_amdgcn_wmma_i32_16x16x64_iu8(false, a0, false, b0, acc00, false, false);
    acc01 = __builtin_amdgcn_wmma_i32_16x16x64_iu8(false, a0, false, b1, acc01, false, false);
    acc10 = __builtin_amdgcn_wmma_i32_16x16x64_iu8(false, a1, false, b0, acc10, false, false);
    acc11 = __builtin_amdgcn_wmma_i32_16x16x64_iu8(false, a1, false, b1, acc11, false, false);
  }
  const int j0 = blockIdx.y * 32 + laneLo, j1 = j0 + 16;
#pragma unroll
  for (int r = 0; r < 8; ++r) {
    int i0 = blockIdx.x * 32 + r + 8 * laneHi;
    int i1 = i0 + 16;
    AdjF[(size_t)i0 * NN + j0] = (acc00[r] > 0 && i0 != j0) ? 1.f : 0.f;
    AdjF[(size_t)i0 * NN + j1] = (acc01[r] > 0 && i0 != j1) ? 1.f : 0.f;
    AdjF[(size_t)i1 * NN + j0] = (acc10[r] > 0 && i1 != j0) ? 1.f : 0.f;
    AdjF[(size_t)i1 * NN + j1] = (acc11[r] > 0 && i1 != j1) ? 1.f : 0.f;
  }
}

__global__ void zero_kernel(float* __restrict__ p, int n)
{
  int i = blockIdx.x * 256 + threadIdx.x;
  if (i < n) p[i] = 0.f;
}

__global__ void tr_cvt_kernel(const float* __restrict__ B, int Kd, int Nn,
                              _Float16* __restrict__ Bt)
{
  int idx = blockIdx.x * 256 + threadIdx.x;
  if (idx < Kd * Nn) {
    int n = idx / Kd, k = idx - n * Kd;
    Bt[idx] = (_Float16)B[(size_t)k * Nn + n];
  }
}

__global__ void node_norm_kernel(const float* __restrict__ X,
                                 float* __restrict__ xhat,
                                 float* __restrict__ xnorm,
                                 float* __restrict__ xext)
{
  int n = blockIdx.x, d = threadIdx.x;
  __shared__ float red[DD];
  __shared__ float sinv;
  float v = X[(size_t)n * DD + d];
  red[d] = v * v;
  __syncthreads();
  for (int s = DD / 2; s > 0; s >>= 1) {
    if (d < s) red[d] += red[d + s];
    __syncthreads();
  }
  if (d == 0) {
    float nrm = sqrtf(red[0]);
    xnorm[n] = nrm;
    sinv = 1.f / fmaxf(nrm, 1e-8f);
  }
  __syncthreads();
  float xh = v * sinv;
  xhat[(size_t)n * DD + d] = xh;
  xext[(size_t)n * DX + d] = xh;
  if (d < DX - DD) xext[(size_t)n * DX + DD + d] = (d == 0) ? 1.f : 0.f;
}

__global__ void att_reduce_kernel(const float* __restrict__ a1,
                                  const float* __restrict__ att_w2,
                                  const float* __restrict__ att_b2,
                                  int k, float* __restrict__ attP)
{
  int tid = threadIdx.x;
  int n = blockIdx.x * 256 + tid;
  const float* w2 = att_w2 + (size_t)k * DH;
  float z = att_b2[k];
  for (int h = 0; h < DH; ++h) z += a1[(size_t)n * DH + h] * w2[h];
  float att = 1.f / (1.f + expf(-z));
  __shared__ float red[256];
  red[tid] = att;
  __syncthreads();
  for (int s = 128; s > 0; s >>= 1) {
    if (tid < s) red[tid] += red[tid + s];
    __syncthreads();
  }
  if (tid == 0) attP[k * 8 + blockIdx.x] = red[0];
}

__global__ void sq_reduce_kernel(const float* __restrict__ src, int count,
                                 float* __restrict__ dst, int slot)
{
  int tid = threadIdx.x;
  float s = 0.f;
  for (int i = blockIdx.x * 256 + tid; i < count; i += gridDim.x * 256) {
    float v = src[i];
    s += v * v;
  }
  __shared__ float red[256];
  red[tid] = s;
  __syncthreads();
  for (int st = 128; st > 0; st >>= 1) {
    if (tid < st) red[tid] += red[tid + st];
    __syncthreads();
  }
  if (tid == 0) dst[slot + blockIdx.x] = red[0];
}

__global__ void gates_kernel(const float* __restrict__ attP,
                             const float* __restrict__ featP,
                             const float* __restrict__ frobP,
                             const int* __restrict__ epoch,
                             float* __restrict__ scal,
                             float* __restrict__ gates_out)
{
  int tid = threadIdx.x;
  __shared__ float simp[KK];
  if (tid < KK) {
    float sa = 0.f, fe = 0.f, fr = 0.f;
    for (int i = 0; i < 8; ++i)  sa += attP[tid * 8 + i];
    for (int i = 0; i < 64; ++i) fe += featP[tid * 64 + i];
    for (int i = 0; i < 8; ++i)  fr += frobP[tid * 8 + i];
    float S_att = sa / (float)NN;
    simp[tid] = BETA_C * sqrtf(fr) * sqrtf(fe) + (1.f - BETA_C) * S_att;
  }
  __syncthreads();
  if (tid == 0) {
    float m = simp[0];
    for (int k = 1; k < KK; ++k) m = fmaxf(m, simp[k]);
    float s = 0.f, pi[KK];
    for (int k = 0; k < KK; ++k) { pi[k] = expf(simp[k] - m); s += pi[k]; }
    float theta_c = TC0_C + (1.f - expf(-LAM_C * (float)epoch[0])) * (TCMAX_C - TC0_C);
    float g[KK];
    for (int k = 0; k < KK; ++k) {
      pi[k] /= s;
      g[k] = clamp01((pi[k] - theta_c) / GEPS + 0.5f);
    }
    int am = 0;
    for (int k = 1; k < KK; ++k) if (g[k] > g[am]) am = k;
    g[am] = fmaxf(g[am], 1.f);
    scal[0] = theta_c;
    for (int k = 0; k < KK; ++k) { scal[1 + k] = g[k]; gates_out[k] = g[k]; }
  }
}

__global__ void edge_scan_kernel(const float* __restrict__ H,
                                 const float* __restrict__ xhat,
                                 float* __restrict__ impE)
{
  const int e = blockIdx.x, k = blockIdx.y, tid = threadIdx.x;
  const float* Hk = H + (size_t)k * NN * EE;
  __shared__ int si0[64], si1[64], sc[64];
  __shared__ float sf[64];
  __shared__ int sij[3];
  int i0 = 0x7FFFFFFF, i1 = 0x7FFFFFFF, cnt = 0;
  for (int n = tid; n < NN; n += 64) {
    if (Hk[(size_t)n * EE + e] > 0.f) {
      ++cnt;
      if (n < i0) { i1 = i0; i0 = n; }
      else if (n < i1) { i1 = n; }
    }
  }
  si0[tid] = i0; si1[tid] = i1; sc[tid] = cnt;
  __syncthreads();
  for (int s = 32; s > 0; s >>= 1) {
    if (tid < s) {
      int a0 = si0[tid], a1 = si1[tid], b0 = si0[tid + s], b1 = si1[tid + s];
      int m0, m1;
      if (a0 <= b0) { m0 = a0; m1 = (a1 < b0) ? a1 : b0; }
      else          { m0 = b0; m1 = (b1 < a0) ? b1 : a0; }
      si0[tid] = m0; si1[tid] = m1; sc[tid] += sc[tid + s];
    }
    __syncthreads();
  }
  if (tid == 0) {
    sij[0] = (si0[0] == 0x7FFFFFFF) ? 0 : si0[0];
    sij[1] = (si1[0] == 0x7FFFFFFF) ? 0 : si1[0];
    sij[2] = sc[0];
  }
  __syncthreads();
  int i = sij[0], j = sij[1];
  float p = xhat[(size_t)i * DD + tid]      * xhat[(size_t)j * DD + tid]
          + xhat[(size_t)i * DD + tid + 64] * xhat[(size_t)j * DD + tid + 64];
  sf[tid] = p;
  __syncthreads();
  for (int s = 32; s > 0; s >>= 1) {
    if (tid < s) sf[tid] += sf[tid + s];
    __syncthreads();
  }
  if (tid == 0) {
    float imp = 0.f;
    if (e < MAXSC) {
      if (sij[2] >= 2) {
        float A = Hk[(size_t)i * EE + e] * Hk[(size_t)j * EE + e];
        imp = A * sf[0];
      } else imp = 0.1f;
    }
    impE[(size_t)k * EE + e] = imp;
  }
}

__global__ void edge_gate_kernel(const float* __restrict__ impE,
                                 const float* __restrict__ scal,
                                 float* __restrict__ eg_out,
                                 float* __restrict__ egEff)
{
  const int k = blockIdx.x, tid = threadIdx.x;
  const float* imp = impE + (size_t)k * EE;
  float* pi = egEff + (size_t)k * EE;
  float* eg = eg_out + (size_t)k * EE;
  __shared__ float sred[256];
  __shared__ int   sidx[256];
  __shared__ float sM, sS, sTheta, sSum;
  __shared__ int   sAmax;

  float m = -1e30f;
  for (int e = tid; e < EE; e += 256) m = fmaxf(m, imp[e]);
  sred[tid] = m; __syncthreads();
  for (int s = 128; s > 0; s >>= 1) { if (tid < s) sred[tid] = fmaxf(sred[tid], sred[tid + s]); __syncthreads(); }
  if (tid == 0) sM = sred[0];
  __syncthreads();

  float sum = 0.f;
  for (int e = tid; e < EE; e += 256) { float p = expf(imp[e] - sM); pi[e] = p; sum += p; }
  sred[tid] = sum; __syncthreads();
  for (int s = 128; s > 0; s >>= 1) { if (tid < s) sred[tid] += sred[tid + s]; __syncthreads(); }
  if (tid == 0) sS = sred[0];
  __syncthreads();

  const float invS = 1.f / sS;
  const float mean = 1.f / (float)EE;
  float vs = 0.f;
  for (int e = tid; e < EE; e += 256) { float d = pi[e] * invS - mean; vs += d * d; }
  sred[tid] = vs; __syncthreads();
  for (int s = 128; s > 0; s >>= 1) { if (tid < s) sred[tid] += sred[tid + s]; __syncthreads(); }
  if (tid == 0) sTheta = fminf(scal[0], mean + sqrtf(sred[0] / (float)(EE - 1)));
  __syncthreads();

  float bv = -1e30f; int bi = EE;
  for (int e = tid; e < EE; e += 256) {
    float p = pi[e];
    if (p > bv || (p == bv && e < bi)) { bv = p; bi = e; }
  }
  sred[tid] = bv; sidx[tid] = bi; __syncthreads();
  for (int s = 128; s > 0; s >>= 1) {
    if (tid < s) {
      if (sred[tid + s] > sred[tid] ||
          (sred[tid + s] == sred[tid] && sidx[tid + s] < sidx[tid])) {
        sred[tid] = sred[tid + s]; sidx[tid] = sidx[tid + s];
      }
    }
    __syncthreads();
  }
  if (tid == 0) sAmax = sidx[0];
  __syncthreads();

  float es = 0.f;
  for (int e = tid; e < EE; e += 256) {
    float g = clamp01((pi[e] * invS - sTheta) / GEPS + 0.5f);
    eg[e] = g;
    es += g;
  }
  sred[tid] = es; __syncthreads();
  for (int s = 128; s > 0; s >>= 1) { if (tid < s) sred[tid] += sred[tid + s]; __syncthreads(); }
  if (tid == 0) {
    sSum = sred[0];
    if (sSum < 1.f) eg[sAmax] = 1.f;
  }
  __syncthreads();

  const float gk = scal[1 + k];
  for (int e = tid; e < EE; e += 256) {
    float g = eg[e];
    egEff[(size_t)k * EE + e] = (gk > 0.5f) ? g : 1.f;
  }
}

__global__ void hp_kernel(const float* __restrict__ H,
                          const float* __restrict__ scal,
                          const float* __restrict__ egEff,
                          float* __restrict__ Hp)
{
  size_t idx = (size_t)blockIdx.x * 256 + threadIdx.x;
  int k = (int)(idx / ((size_t)NN * EE));
  int e = (int)(idx % EE);
  Hp[idx] = H[idx] * scal[1 + k] * egEff[(size_t)k * EE + e];
}

__global__ void pack_kernel(const float* __restrict__ Hp_k,
                            unsigned char* __restrict__ Bi8)
{
  size_t i = (size_t)blockIdx.x * 256 + threadIdx.x;
  Bi8[i] = Hp_k[i] > 0.f ? (unsigned char)1 : (unsigned char)0;
}

__global__ void npi_kernel(const float* __restrict__ xhat,
                           const float* __restrict__ xnorm,
                           const float* __restrict__ nsum,
                           float* __restrict__ sbuf,
                           float* __restrict__ npi_out)
{
  const int tid = threadIdx.x;
  __shared__ float sred[256];
  __shared__ float sM, sS;
  for (int n = tid; n < NN; n += 256) {
    float deg = nsum[(size_t)n * DX + DD];
    float dot = 0.f;
    for (int d = 0; d < DD; ++d) dot += xhat[(size_t)n * DD + d] * nsum[(size_t)n * DX + d];
    float avg = (deg > 0.f) ? dot / fmaxf(deg, 1.f) : 0.f;
    sbuf[n] = avg * xnorm[n];
  }
  __syncthreads();
  float m = -1e30f;
  for (int n = tid; n < NN; n += 256) m = fmaxf(m, sbuf[n]);
  sred[tid] = m; __syncthreads();
  for (int s = 128; s > 0; s >>= 1) { if (tid < s) sred[tid] = fmaxf(sred[tid], sred[tid + s]); __syncthreads(); }
  if (tid == 0) sM = sred[0];
  __syncthreads();
  float sum = 0.f;
  for (int n = tid; n < NN; n += 256) sum += expf(sbuf[n] - sM);
  sred[tid] = sum; __syncthreads();
  for (int s = 128; s > 0; s >>= 1) { if (tid < s) sred[tid] += sred[tid + s]; __syncthreads(); }
  if (tid == 0) sS = sred[0];
  __syncthreads();
  for (int n = tid; n < NN; n += 256) npi_out[n] = expf(sbuf[n] - sM) / sS;
}

extern "C" void kernel_launch(void* const* d_in, const int* in_sizes, int n_in,
                              void* d_out, int out_size, void* d_ws, size_t ws_size,
                              hipStream_t stream)
{
  const float* H       = (const float*)d_in[0];
  const float* X       = (const float*)d_in[1];
  const int*   epoch   = (const int*)d_in[2];
  const float* mlp_w1  = (const float*)d_in[3];
  const float* mlp_b1  = (const float*)d_in[4];
  const float* mlp_w2  = (const float*)d_in[5];
  const float* mlp_b2  = (const float*)d_in[6];
  const float* comp_W  = (const float*)d_in[7];
  const float* att_w1  = (const float*)d_in[8];
  const float* att_b1  = (const float*)d_in[9];
  const float* att_w2  = (const float*)d_in[10];
  const float* att_b2  = (const float*)d_in[11];

  float* out       = (float*)d_out;
  float* Hp        = out;
  float* gates_out = out + (size_t)KK * NN * EE;
  float* eg_out    = gates_out + KK;
  float* npi_out   = eg_out + (size_t)KK * EE;

  float* ws    = (float*)d_ws;
  float* Xk    = ws;
  float* h1b   = Xk    + (size_t)KK * NN * DD;
  float* xhat  = h1b   + (size_t)NN * DD;
  float* xnorm = xhat  + (size_t)NN * DD;
  float* xext  = xnorm + NN;
  float* impE  = xext  + (size_t)NN * DX;
  float* egEff = impE  + (size_t)KK * EE;
  float* scal  = egEff + (size_t)KK * EE;
  float* zbias = scal  + 16;
  float* attP  = zbias + 256;
  float* featP = attP  + KK * 8;
  float* frobP = featP + KK * 64;
  float* AdjF  = frobP + KK * 8;
  float* nsum  = AdjF  + (size_t)NN * NN;
  float* sbuf  = nsum  + (size_t)NN * DX;
  _Float16* w1t   = (_Float16*)(sbuf + NN);
  _Float16* w2t   = w1t + DD * DD;
  _Float16* awt   = w2t + DD * DD;
  _Float16* xextT = awt + DD * DH;
  unsigned char* Bi8 = (unsigned char*)(xextT + (size_t)DX * NN);

  zero_kernel<<<dim3(1), dim3(256), 0, stream>>>(zbias, 256);
  node_norm_kernel<<<dim3(NN), dim3(DD), 0, stream>>>(X, xhat, xnorm, xext);
  tr_cvt_kernel<<<dim3((NN * DX + 255) / 256), dim3(256), 0, stream>>>(xext, NN, DX, xextT);

  for (int k = 0; k < KK; ++k) {
    tr_cvt_kernel<<<dim3((DD * DD + 255) / 256), dim3(256), 0, stream>>>(
        mlp_w1 + (size_t)k * DD * DD, DD, DD, w1t);
    gemm_f16_wmma<<<dim3(NN / 32, DD / 32), dim3(32), 0, stream>>>(
        X, DD, w1t, mlp_b1 + (size_t)k * DD, h1b, DD, DD, 1);
    tr_cvt_kernel<<<dim3((DD * DD + 255) / 256), dim3(256), 0, stream>>>(
        mlp_w2 + (size_t)k * DD * DD, DD, DD, w2t);
    gemm_f16_wmma<<<dim3(NN / 32, DD / 32), dim3(32), 0, stream>>>(
        h1b, DD, w2t, mlp_b2 + (size_t)k * DD, Xk + (size_t)k * NN * DD, DD, DD, 0);
    tr_cvt_kernel<<<dim3((DD * DH + 255) / 256), dim3(256), 0, stream>>>(
        att_w1 + (size_t)k * DD * DH, DD, DH, awt);
    gemm_f16_wmma<<<dim3(NN / 32, DH / 32), dim3(32), 0, stream>>>(
        Xk + (size_t)k * NN * DD, DD, awt, att_b1 + (size_t)k * DH, h1b, DH, DD, 1);
    att_reduce_kernel<<<dim3(8), dim3(256), 0, stream>>>(h1b, att_w2, att_b2, k, attP);
    sq_reduce_kernel<<<dim3(64), dim3(256), 0, stream>>>(
        Xk + (size_t)k * NN * DD, NN * DD, featP, k * 64);
    sq_reduce_kernel<<<dim3(8), dim3(256), 0, stream>>>(
        comp_W + (size_t)k * DD * DD, DD * DD, frobP, k * 8);
  }

  gates_kernel<<<dim3(1), dim3(32), 0, stream>>>(attP, featP, frobP, epoch, scal, gates_out);

  edge_scan_kernel<<<dim3(EE, KK), dim3(64), 0, stream>>>(H, xhat, impE);
  edge_gate_kernel<<<dim3(KK), dim3(256), 0, stream>>>(impE, scal, eg_out, egEff);

  hp_kernel<<<dim3((KK * NN * EE) / 256), dim3(256), 0, stream>>>(H, scal, egEff, Hp);

  for (int k = 0; k < KK; ++k) {
    pack_kernel<<<dim3((NN * EE) / 256), dim3(256), 0, stream>>>(
        Hp + (size_t)k * NN * EE, Bi8);
    adj_wmma_iu8<<<dim3(NN / 32, NN / 32), dim3(32), 0, stream>>>(Bi8, AdjF);
    gemm_f16_wmma<<<dim3(NN / 32, DX / 32), dim3(32), 0, stream>>>(
        AdjF, NN, xextT, zbias, nsum, DX, NN, 0);
    npi_kernel<<<dim3(1), dim3(256), 0, stream>>>(
        xhat, xnorm, nsum, sbuf, npi_out + (size_t)k * NN);
  }
}
